// GCNNet_25460566130736
// MI455X (gfx1250) — matrix operand order repaired and checked by reading, not compiled, not run on hardware
//
#include <hip/hip_runtime.h>
#include <hip/hip_bf16.h>

// GatedGCN (4 layers, D=128) for MI455X gfx1250.
// GEMMs via v_wmma_f32_16x16x32_bf16 (wave32, fp32 accum). Memory-bound
// workload: edge tensors stream from HBM, node tensors live in 192MB L2.

#define NN 40000
#define NE 640000
#define DD 128
#define NLAYERS 4

typedef __attribute__((ext_vector_type(16))) __bf16 v16bf;
typedef __attribute__((ext_vector_type(8)))  float  v8f;

__device__ __forceinline__ v8f wmma_bf16(v16bf a, v16bf b, v8f c) {
  // 8 args: (neg_a, A, neg_b, B, c_mod, C, reuse_a, reuse_b)
  return __builtin_amdgcn_wmma_f32_16x16x32_bf16(false, a, false, b, (short)0, c,
                                                 false, false);
}

// ---------------------------------------------------------------------------
// A-fragment (16x32 bf16, ISA 7.12.2): lane<16 holds row M=lane, elements
// 0..7 -> K=kbase+0..7, 8..15 -> K=kbase+16..23; lane>=16 same row, +8 offset.
// ---------------------------------------------------------------------------
__device__ __forceinline__ v16bf load_a_frag(const float* __restrict__ rowptr,
                                             int kbase, int koff) {
  float4 q0 = *(const float4*)(rowptr + kbase + koff);
  float4 q1 = *(const float4*)(rowptr + kbase + koff + 4);
  float4 q2 = *(const float4*)(rowptr + kbase + koff + 16);
  float4 q3 = *(const float4*)(rowptr + kbase + koff + 20);
  v16bf a;
  a[0]  = (__bf16)q0.x; a[1]  = (__bf16)q0.y; a[2]  = (__bf16)q0.z; a[3]  = (__bf16)q0.w;
  a[4]  = (__bf16)q1.x; a[5]  = (__bf16)q1.y; a[6]  = (__bf16)q1.z; a[7]  = (__bf16)q1.w;
  a[8]  = (__bf16)q2.x; a[9]  = (__bf16)q2.y; a[10] = (__bf16)q2.z; a[11] = (__bf16)q2.w;
  a[12] = (__bf16)q3.x; a[13] = (__bf16)q3.y; a[14] = (__bf16)q3.z; a[15] = (__bf16)q3.w;
  return a;
}

// B-fragments are pre-swizzled in global memory: contiguous 16 bf16 per lane.
__device__ __forceinline__ v16bf load_b_frag(const __bf16* __restrict__ wf,
                                             int nt, int kc, int lane) {
  return *(const v16bf*)(wf + ((((nt * 4 + kc) * 32) + lane) << 4));
}

// ---------------------------------------------------------------------------
// Pre-swizzle one fp32 128x128 weight into bf16 B-fragment order.
// frag layout within a matrix: [nt:8][kc:4][lane:32][i:16]
// element (lane,i) of chunk kc,nt = W[kc*32 + (lane>=16?8:0) + (i<8?i:i+8)][nt*16 + (lane&15)]
// ---------------------------------------------------------------------------
__global__ __launch_bounds__(256) void prep_weights(const float* __restrict__ W,
                                                    __bf16* __restrict__ wfrag) {
  int idx = blockIdx.x * 256 + threadIdx.x;
  if (idx >= 20 * 16384) return;
  int mat  = idx >> 14;
  int t    = idx & 16383;
  int i    = t & 15;
  int lane = (t >> 4) & 31;
  int kc   = (t >> 9) & 3;
  int nt   = (t >> 11) & 7;
  int k = kc * 32 + ((lane & 16) ? 8 : 0) + (i < 8 ? i : i + 8);
  int n = nt * 16 + (lane & 15);
  wfrag[idx] = (__bf16)W[((size_t)mat * DD + k) * DD + n];
}

// One 16-row strip x full 128 cols against one weight matrix; A regs reused.
__device__ __forceinline__ void gemm_one(v16bf a0, v16bf a1, v16bf a2, v16bf a3,
                                         const __bf16* __restrict__ wf,
                                         const float* __restrict__ bp,
                                         float* __restrict__ outp,
                                         int mbase, int lane) {
  int col    = lane & 15;
  int rowoff = (lane & 16) ? 8 : 0;
#pragma unroll
  for (int nt = 0; nt < 8; ++nt) {
    v8f c = {0.f, 0.f, 0.f, 0.f, 0.f, 0.f, 0.f, 0.f};
    c = wmma_bf16(a0, load_b_frag(wf, nt, 0, lane), c);
    c = wmma_bf16(a1, load_b_frag(wf, nt, 1, lane), c);
    c = wmma_bf16(a2, load_b_frag(wf, nt, 2, lane), c);
    c = wmma_bf16(a3, load_b_frag(wf, nt, 3, lane), c);
    float bv = bp[nt * 16 + col];
#pragma unroll
    for (int r = 0; r < 8; ++r)
      outp[(size_t)(mbase + rowoff + r) * DD + nt * 16 + col] = c[r] + bv;
  }
}

// Node GEMMs: Ah, Bh, Dh, Eh from one load of the h tile (4x weight reuse).
__global__ __launch_bounds__(256) void node_gemm(const float* __restrict__ h,
                                                 const __bf16* __restrict__ wfrag,
                                                 const float* __restrict__ bias,
                                                 float* __restrict__ Ah,
                                                 float* __restrict__ Bh,
                                                 float* __restrict__ Dh,
                                                 float* __restrict__ Eh,
                                                 int ntiles) {
  int wave = threadIdx.x >> 5;
  int lane = threadIdx.x & 31;
  int tile = blockIdx.x * 8 + wave;
  if (tile >= ntiles) return;  // whole-wave guard: EXEC stays all-ones
  int mbase = tile * 16;
  const float* rowptr = h + (size_t)(mbase + (lane & 15)) * DD;
  int koff = (lane & 16) ? 8 : 0;
  v16bf a0 = load_a_frag(rowptr, 0,  koff);
  v16bf a1 = load_a_frag(rowptr, 32, koff);
  v16bf a2 = load_a_frag(rowptr, 64, koff);
  v16bf a3 = load_a_frag(rowptr, 96, koff);
  gemm_one(a0, a1, a2, a3, wfrag + 0 * 16384, bias + 0 * 128, Ah, mbase, lane);
  gemm_one(a0, a1, a2, a3, wfrag + 1 * 16384, bias + 1 * 128, Bh, mbase, lane);
  gemm_one(a0, a1, a2, a3, wfrag + 3 * 16384, bias + 3 * 128, Dh, mbase, lane);
  gemm_one(a0, a1, a2, a3, wfrag + 4 * 16384, bias + 4 * 128, Eh, mbase, lane);
}

// Edge GEMM fused with gather + sigmoid + gated atomic scatter.
// e_pre = e@Wc + bc + Dh[src] + Eh[dst]; num[dst]+=sig*Bh[src]; den[dst]+=sig
__global__ __launch_bounds__(256) void edge_gemm(const float* __restrict__ e,
                                                 const __bf16* __restrict__ wf,
                                                 const float* __restrict__ bp,
                                                 const float* __restrict__ Bh,
                                                 const float* __restrict__ Dh,
                                                 const float* __restrict__ Eh,
                                                 const int* __restrict__ src,
                                                 const int* __restrict__ dst,
                                                 float* __restrict__ e_pre,
                                                 float* __restrict__ num,
                                                 float* __restrict__ den,
                                                 int ntiles) {
  int wave = threadIdx.x >> 5;
  int lane = threadIdx.x & 31;
  int tile = blockIdx.x * 8 + wave;
  if (tile >= ntiles) return;
  int mbase = tile * 16;
  const float* rowptr = e + (size_t)(mbase + (lane & 15)) * DD;
  int koff = (lane & 16) ? 8 : 0;
  v16bf a0 = load_a_frag(rowptr, 0,  koff);
  v16bf a1 = load_a_frag(rowptr, 32, koff);
  v16bf a2 = load_a_frag(rowptr, 64, koff);
  v16bf a3 = load_a_frag(rowptr, 96, koff);
  int coll   = lane & 15;
  int rowoff = (lane & 16) ? 8 : 0;
#pragma unroll
  for (int nt = 0; nt < 8; ++nt) {
    v8f c = {0.f, 0.f, 0.f, 0.f, 0.f, 0.f, 0.f, 0.f};
    c = wmma_bf16(a0, load_b_frag(wf, nt, 0, lane), c);
    c = wmma_bf16(a1, load_b_frag(wf, nt, 1, lane), c);
    c = wmma_bf16(a2, load_b_frag(wf, nt, 2, lane), c);
    c = wmma_bf16(a3, load_b_frag(wf, nt, 3, lane), c);
    int   col = nt * 16 + coll;
    float bv  = bp[col];
#pragma unroll
    for (int r = 0; r < 8; ++r) {
      int eidx = mbase + rowoff + r;
      int s = src[eidx];
      int d = dst[eidx];
      float v = c[r] + bv + Dh[(size_t)s * DD + col] + Eh[(size_t)d * DD + col];
      e_pre[(size_t)eidx * DD + col] = v;
      float sg = 1.0f / (1.0f + __expf(-v));
      atomicAdd(num + (size_t)d * DD + col, sg * Bh[(size_t)s * DD + col]);
      atomicAdd(den + (size_t)d * DD + col, sg);
    }
  }
}

// h_pre = Ah + num/(den+eps)   (in place over Ah)
__global__ __launch_bounds__(256) void node_combine(float* __restrict__ Ah,
                                                    const float* __restrict__ num,
                                                    const float* __restrict__ den,
                                                    size_t total) {
  size_t idx = (size_t)blockIdx.x * 256 + threadIdx.x;
  if (idx >= total) return;
  Ah[idx] = Ah[idx] + num[idx] / (den[idx] + 1e-6f);
}

// Per-column sum & sumsq; LDS reduce, few global atomics. acc = [sum128|sq128]
__global__ __launch_bounds__(256) void bn_stats(const float* __restrict__ x,
                                                int rows, float* __restrict__ acc) {
  int g = blockIdx.x * 256 + threadIdx.x;
  int n = g & 127;
  int w = g >> 7;
  int nw = (gridDim.x * 256) >> 7;
  float s = 0.f, s2 = 0.f;
  for (int r = w; r < rows; r += nw) {
    float v = x[(size_t)r * DD + n];
    s += v;
    s2 += v * v;
  }
  __shared__ float sh[512];
  sh[threadIdx.x]       = s;
  sh[256 + threadIdx.x] = s2;
  __syncthreads();
  if (threadIdx.x < 128) {
    s  = sh[threadIdx.x] + sh[threadIdx.x + 128];
    s2 = sh[256 + threadIdx.x] + sh[256 + threadIdx.x + 128];
    atomicAdd(acc + threadIdx.x, s);
    atomicAdd(acc + 128 + threadIdx.x, s2);
  }
}

// scale = gamma*rsqrt(var+eps); shift = beta - mu*scale
__global__ void bn_finalize(const float* __restrict__ acc, float cnt_inv,
                            const float* __restrict__ gamma,
                            const float* __restrict__ beta,
                            float* __restrict__ scale, float* __restrict__ shift) {
  int n = threadIdx.x;  // 128 threads
  float mu  = acc[n] * cnt_inv;
  float var = acc[128 + n] * cnt_inv - mu * mu;
  float sc  = gamma[n] * rsqrtf(var + 1e-5f);
  scale[n] = sc;
  shift[n] = beta[n] - mu * sc;
}

// outp = cur + relu(pre*scale + shift)   (outp may alias cur or pre)
__global__ __launch_bounds__(256) void apply_residual(const float* __restrict__ cur,
                                                      const float* __restrict__ pre,
                                                      const float* __restrict__ scale,
                                                      const float* __restrict__ shift,
                                                      float* __restrict__ outp,
                                                      size_t total) {
  size_t idx = (size_t)blockIdx.x * 256 + threadIdx.x;
  if (idx >= total) return;
  int col = (int)(idx & 127);
  float v = pre[idx] * scale[col] + shift[col];
  v = v > 0.f ? v : 0.f;
  outp[idx] = cur[idx] + v;
}

extern "C" void kernel_launch(void* const* d_in, const int* in_sizes, int n_in,
                              void* d_out, int out_size, void* d_ws, size_t ws_size,
                              hipStream_t stream) {
  (void)in_sizes; (void)n_in; (void)out_size; (void)ws_size;
  const float* h     = (const float*)d_in[0];
  const float* e     = (const float*)d_in[1];
  const float* W     = (const float*)d_in[2];
  const float* bias  = (const float*)d_in[3];
  const float* gamma = (const float*)d_in[4];
  const float* beta  = (const float*)d_in[5];
  const int*   src   = (const int*)d_in[6];
  const int*   dst   = (const int*)d_in[7];

  float* out   = (float*)d_out;
  float* out_h = out;                       // copy of input h
  float* out_e = out + (size_t)NN * DD;     // also doubles as e_pre scratch

  char*  base = (char*)d_ws;
  size_t off  = 0;
  auto alloc = [&](size_t bytes) -> void* {
    void* p = base + off;
    off += (bytes + 255) & ~(size_t)255;
    return p;
  };
  float*  h_cur   = (float*)alloc((size_t)NN * DD * 4);
  float*  e_cur   = (float*)alloc((size_t)NE * DD * 4);
  float*  Ah      = (float*)alloc((size_t)NN * DD * 4);  // becomes h_pre
  float*  Bh      = (float*)alloc((size_t)NN * DD * 4);
  float*  Dh      = (float*)alloc((size_t)NN * DD * 4);
  float*  Eh      = (float*)alloc((size_t)NN * DD * 4);
  float*  num     = (float*)alloc((size_t)NN * DD * 4);
  float*  den     = (float*)alloc((size_t)NN * DD * 4);
  float*  acc_h   = (float*)alloc(256 * 4);
  float*  acc_e   = (float*)alloc(256 * 4);
  float*  scale_h = (float*)alloc(128 * 4);
  float*  shift_h = (float*)alloc(128 * 4);
  float*  scale_e = (float*)alloc(128 * 4);
  float*  shift_e = (float*)alloc(128 * 4);
  __bf16* wfrag   = (__bf16*)alloc((size_t)20 * 16384 * 2);

  hipMemcpyAsync(h_cur, h, (size_t)NN * DD * 4, hipMemcpyDeviceToDevice, stream);
  hipMemcpyAsync(out_h, h, (size_t)NN * DD * 4, hipMemcpyDeviceToDevice, stream);
  hipMemcpyAsync(e_cur, e, (size_t)NE * DD * 4, hipMemcpyDeviceToDevice, stream);

  prep_weights<<<1280, 256, 0, stream>>>(W, wfrag);

  for (int l = 0; l < NLAYERS; ++l) {
    hipMemsetAsync(num, 0, (size_t)NN * DD * 4, stream);
    hipMemsetAsync(den, 0, (size_t)NN * DD * 4, stream);
    hipMemsetAsync(acc_h, 0, 256 * 4, stream);
    hipMemsetAsync(acc_e, 0, 256 * 4, stream);

    node_gemm<<<(2500 + 7) / 8, 256, 0, stream>>>(
        h_cur, wfrag + (size_t)l * 5 * 16384, bias + (size_t)l * 5 * 128,
        Ah, Bh, Dh, Eh, 2500);
    edge_gemm<<<40000 / 8, 256, 0, stream>>>(
        e_cur, wfrag + ((size_t)l * 5 + 2) * 16384, bias + ((size_t)l * 5 + 2) * 128,
        Bh, Dh, Eh, src, dst, out_e, num, den, 40000);
    node_combine<<<20000, 256, 0, stream>>>(Ah, num, den, (size_t)NN * DD);
    bn_stats<<<512, 256, 0, stream>>>(Ah, NN, acc_h);
    bn_stats<<<512, 256, 0, stream>>>(out_e, NE, acc_e);
    bn_finalize<<<1, 128, 0, stream>>>(acc_h, 1.0f / NN,
                                       gamma + (size_t)(l * 2 + 0) * 128,
                                       beta + (size_t)(l * 2 + 0) * 128,
                                       scale_h, shift_h);
    bn_finalize<<<1, 128, 0, stream>>>(acc_e, 1.0f / NE,
                                       gamma + (size_t)(l * 2 + 1) * 128,
                                       beta + (size_t)(l * 2 + 1) * 128,
                                       scale_e, shift_e);
    apply_residual<<<20000, 256, 0, stream>>>(h_cur, Ah, scale_h, shift_h, h_cur,
                                              (size_t)NN * DD);
    float* edstp = (l == NLAYERS - 1) ? out_e : e_cur;
    apply_residual<<<320000, 256, 0, stream>>>(e_cur, out_e, scale_e, shift_e, edstp,
                                               (size_t)NE * DD);
  }
}
